// Dual_Mamba_84413287235930
// MI455X (gfx1250) — matrix operand order repaired and checked by reading, not compiled, not run on hardware
//
#include <hip/hip_runtime.h>
#include <hip/hip_bf16.h>
#include <math.h>

// ---------------- problem constants (from reference) ----------------
#define BSZc   4
#define T0c    8
#define SEQ    2048            // T0*H0*W0
#define DM     1024            // D_MODEL
#define DI     2048            // D_INNER
#define DI2    4096            // 2*D_INNER
#define DTR    64              // DT_RANK
#define DST    16              // D_STATE
#define XDE    96              // DTR + 2*DST
#define BL     (BSZc*SEQ)      // 8192 rows for all GEMMs
#define OUT_ELEMS ((size_t)BL*DM)   // 8388608, then +1 for align_loss

typedef __bf16 bf16_t;
typedef __attribute__((ext_vector_type(16))) __bf16 bf16x16;
typedef __attribute__((ext_vector_type(8)))  __bf16 bf16x8;
typedef __attribute__((ext_vector_type(8)))  float  v8f;

union bfrag { bf16x16 f; bf16x8 h[2]; };

// Build a WMMA fragment from two contiguous 16-byte chunks (-> 2x global_load_b128).
__device__ inline bf16x16 load_frag2(const bf16_t* __restrict__ p0,
                                     const bf16_t* __restrict__ p1) {
  bfrag u;
  u.h[0] = *(const bf16x8*)p0;
  u.h[1] = *(const bf16x8*)p1;
  return u.f;
}

// Branch-free numerically-stable softplus (no EXEC-modifying cmpx near WMMA).
__device__ inline float softplus_bf(float v) {
  return fmaxf(v, 0.f) + log1pf(__expf(-fabsf(v)));
}

// ---------------- generic C = A(MxK) * W(NxK)^T, bf16 in / f32 out ----------------
// A-fragment per lane: rows m=lane&15; elements = K[kb..kb+7] ++ K[kb+16..kb+23],
//   kb = (lane&16)?8:0  -> two aligned 16B runs.
// B-fragment per lane: col n=lane&15; elements = K[bkb..bkb+7] ++ K[bkb+16..bkb+23],
//   bkb=(lane&16)?8:0 -> two aligned 16B runs.
// NT = 16-wide N tiles per wave (A fragment reused NT times, NT independent accs).
template <int NT>
__global__ void gemm_bf16(const bf16_t* __restrict__ A, const bf16_t* __restrict__ W,
                          float* __restrict__ C, int M, int N, int K) {
  const int lane = threadIdx.x & 31;
  const int wave = blockIdx.x * (blockDim.x >> 5) + (threadIdx.x >> 5);
  const int ntg  = N / (16 * NT);
  const int tm   = (wave / ntg) << 4;
  const int tn0  = (wave % ntg) * (16 * NT);
  if (tm >= M) return;                         // wave-uniform
  const int akb = (lane & 16) ? 8 : 0;
  const int bkb = (lane & 16) ? 8 : 0;
  const bf16_t* Ap = A + ((size_t)tm + (lane & 15)) * K + akb;
  const bf16_t* Wp[NT];
#pragma unroll
  for (int i = 0; i < NT; ++i)
    Wp[i] = W + ((size_t)(tn0 + i * 16) + (lane & 15)) * K + bkb;
  v8f acc[NT];
#pragma unroll
  for (int i = 0; i < NT; ++i) acc[i] = {};
  for (int k0 = 0; k0 < K; k0 += 32) {
    __builtin_prefetch(Ap + k0 + 256, 0, 1);   // global_prefetch_b8
    bf16x16 a = load_frag2(Ap + k0, Ap + k0 + 16);
#pragma unroll
    for (int i = 0; i < NT; ++i) {
      bf16x16 b = load_frag2(Wp[i] + k0, Wp[i] + k0 + 16);
      acc[i] = __builtin_amdgcn_wmma_f32_16x16x32_bf16(false, a, false, b, (short)0,
                                                       acc[i], false, false);
    }
  }
  const int n  = lane & 15;
  const int mb = (lane & 16) ? 8 : 0;
#pragma unroll
  for (int i = 0; i < NT; ++i) {
    const int tn = tn0 + i * 16;
#pragma unroll
    for (int r = 0; r < 8; ++r)
      C[(size_t)(tm + mb + r) * N + tn + n] = acc[i][r];
  }
}

// ---------------- fp32 -> bf16 convert ----------------
__global__ void f32_to_bf16(const float* __restrict__ src, bf16_t* __restrict__ dst,
                            size_t n) {
  size_t i = (size_t)blockIdx.x * blockDim.x + threadIdx.x;
  if (i < n) dst[i] = (bf16_t)src[i];
}

// pack dt_low = x_dbl[:, 0:64] into contiguous bf16 (BL x 64)
__global__ void pack_dtlow(const float* __restrict__ x_dbl, bf16_t* __restrict__ dtlow) {
  int i = blockIdx.x * blockDim.x + threadIdx.x;   // over BL*DTR
  if (i >= BL * DTR) return;
  int r = i >> 6, c = i & 63;
  dtlow[i] = (bf16_t)x_dbl[(size_t)r * XDE + c];
}

// ---------------- channel alignment: direct 2D DFT + band energies ----------------
__global__ void fft_band_kernel(const float* __restrict__ xz, float* __restrict__ features) {
  const int c  = blockIdx.x;                 // 0..2047
  const int bt = blockIdx.y;                 // 0..31
  const int b  = bt / T0c;
  const int t  = bt % T0c;
  __shared__ float sig[256];
  __shared__ float cost[16], sint[16];
  __shared__ float bands[8];
  const int tid = threadIdx.x;               // 64 threads
  if (tid < 8) bands[tid] = 0.f;
  if (tid < 16) {
    float ang = -6.2831853071795864f * (float)tid / 16.f;
    cost[tid] = __cosf(ang);
    sint[tid] = __sinf(ang);
  }
  for (int i = tid; i < 256; i += 64) {
    int l = t * 256 + i;
    sig[i] = xz[((size_t)(b * SEQ + l)) * DI2 + c];
  }
  __syncthreads();
  for (int bin = tid; bin < 16 * 9; bin += 64) {   // rfft2 bins: kh 0..15, kw 0..8
    const int kh = bin / 9, kw = bin % 9;
    float re = 0.f, im = 0.f;
#pragma unroll 1
    for (int h = 0; h < 16; ++h) {
      const int ph = (kh * h) & 15;
#pragma unroll
      for (int w = 0; w < 16; ++w) {
        const int p = (ph + kw * w) & 15;
        const float v = sig[h * 16 + w];
        re += v * cost[p];
        im += v * sint[p];
      }
    }
    float mag = sqrtf(re * re + im * im + 1e-8f);
    int hs = (kh + 8) & 15;                  // fftshift
    int ws = (kw + 4) % 9;
    float dy = (float)(hs - 8) / 16.f;
    float dx = (float)(ws - 4) / 9.f;
    float r  = sqrtf(dy * dy + dx * dx);
    int band = (int)(r * 8.f);
    if (band > 7) band = 7;
    atomicAdd(&bands[band], mag);
  }
  __syncthreads();
  if (tid == 0) {
    float tot = 1e-8f;
    for (int k = 0; k < 8; ++k) tot += bands[k];
    features[bt * DI + c] = (bands[4] + bands[5] + bands[6] + bands[7]) / tot;
  }
}

// ---------------- align loss + A inversion (single block) ----------------
// sim.sum() == || sum_i fn_i ||^2 ; loss = 1 - that/1024.
__global__ void stats_invertA_kernel(const float* __restrict__ features,
                                     const float* __restrict__ A_log,
                                     float* __restrict__ Anew,
                                     float* __restrict__ loss_out,
                                     float* __restrict__ scratch) {  // 3*2048 floats
  __shared__ float red[256];
  __shared__ float rn[32];
  __shared__ float sAmax[16], sAmin[16];
  __shared__ float s_maxm, s_sumexp, s_attnmax, s_sumg2;
  const int tid = threadIdx.x;
  for (int i = 0; i < 32; ++i) {             // row norms of features (32 x 2048)
    float p = 0.f;
    for (int c = tid; c < DI; c += 256) { float v = features[i * DI + c]; p += v * v; }
    red[tid] = p; __syncthreads();
    for (int s = 128; s > 0; s >>= 1) { if (tid < s) red[tid] += red[tid + s]; __syncthreads(); }
    if (tid == 0) rn[i] = fmaxf(sqrtf(red[0]), 1e-12f);
    __syncthreads();
  }
  float psumg2 = 0.f, pmax = -1e30f;
  for (int c = tid; c < DI; c += 256) {
    float g = 0.f, m = 0.f;
    for (int i = 0; i < 32; ++i) { float v = features[i * DI + c]; g += v / rn[i]; m += v; }
    m *= (1.f / 32.f);
    scratch[c] = m;
    psumg2 += g * g;
    pmax = fmaxf(pmax, m);
  }
  red[tid] = psumg2; __syncthreads();
  for (int s = 128; s > 0; s >>= 1) { if (tid < s) red[tid] += red[tid + s]; __syncthreads(); }
  if (tid == 0) s_sumg2 = red[0]; __syncthreads();
  red[tid] = pmax; __syncthreads();
  for (int s = 128; s > 0; s >>= 1) { if (tid < s) red[tid] = fmaxf(red[tid], red[tid + s]); __syncthreads(); }
  if (tid == 0) s_maxm = red[0]; __syncthreads();
  float pse = 0.f;
  for (int c = tid; c < DI; c += 256) {
    float e = __expf(scratch[c] - s_maxm);
    scratch[DI + c] = e;
    pse += e;
  }
  red[tid] = pse; __syncthreads();
  for (int s = 128; s > 0; s >>= 1) { if (tid < s) red[tid] += red[tid + s]; __syncthreads(); }
  if (tid == 0) s_sumexp = red[0]; __syncthreads();
  float pamax = 0.f;
  for (int d = tid; d < DI; d += 256) {
    float acc = 0.f;
#pragma unroll
    for (int n = 0; n < DST; ++n) { float e = __expf(A_log[d * DST + n]); acc += e * e; }
    float a = sqrtf(acc);
    scratch[2 * DI + d] = a;
    pamax = fmaxf(pamax, a);
  }
  red[tid] = pamax; __syncthreads();
  for (int s = 128; s > 0; s >>= 1) { if (tid < s) red[tid] = fmaxf(red[tid], red[tid + s]); __syncthreads(); }
  if (tid == 0) s_attnmax = red[0]; __syncthreads();
  if (tid < DST) {                           // per-column A max/min
    float mx = -1e30f, mn = 1e30f;
    for (int d = 0; d < DI; ++d) { float v = A_log[d * DST + tid]; mx = fmaxf(mx, v); mn = fminf(mn, v); }
    sAmax[tid] = mx; sAmin[tid] = mn;
  }
  __syncthreads();
  for (int d = tid; d < DI; d += 256) {      // blended A, then A = -exp(.)
    float f    = scratch[DI + d] / s_sumexp;
    float attn = scratch[2 * DI + d] / (s_attnmax + 1e-8f);
    float alpha = fminf(fmaxf(f * (1.f - attn), 0.f), 1.f);
#pragma unroll
    for (int n = 0; n < DST; ++n) {
      float al = A_log[d * DST + n];
      float fl = sAmax[n] + sAmin[n] - al;
      Anew[d * DST + n] = -__expf((1.f - alpha) * al + alpha * fl);
    }
  }
  if (tid == 0) *loss_out = 1.f - s_sumg2 * (1.f / 1024.f);
}

// ---------------- depthwise causal conv (K=4) + SiLU ----------------
__global__ void conv_silu_kernel(const float* __restrict__ xz,
                                 const float* __restrict__ conv_w,
                                 const float* __restrict__ conv_b,
                                 float* __restrict__ xc, bf16_t* __restrict__ xc_bf) {
  const int idx = blockIdx.x * blockDim.x + threadIdx.x;  // over BL*DI
  if (idx >= BL * DI) return;
  const int d  = idx & (DI - 1);
  const int bl = idx >> 11;
  const int l  = bl & (SEQ - 1);
  const int b  = bl >> 11;
  float acc = conv_b[d];
#pragma unroll
  for (int k = 0; k < 4; ++k) {
    int ls = l - 3 + k;
    if (ls >= 0) acc += conv_w[d * 4 + k] * xz[((size_t)(b * SEQ + ls)) * DI2 + d];
  }
  float s = acc / (1.f + __expf(-acc));      // SiLU
  xc[(size_t)bl * DI + d]    = s;
  xc_bf[(size_t)bl * DI + d] = (bf16_t)s;
}

// ---------------- dt_proj GEMM (K=64, packed dt_low) + bias + softplus ----------------
__global__ void dt_gemm_softplus(const bf16_t* __restrict__ dtlow,
                                 const bf16_t* __restrict__ dtw_bf,
                                 const float* __restrict__ dt_b,
                                 float* __restrict__ delta) {
  constexpr int NT = 4;
  const int lane = threadIdx.x & 31;
  const int wave = blockIdx.x * (blockDim.x >> 5) + (threadIdx.x >> 5);
  const int ntg  = DI / (16 * NT);           // 32
  const int tm   = (wave / ntg) << 4;
  const int tn0  = (wave % ntg) * (16 * NT);
  if (tm >= BL) return;
  const int akb = (lane & 16) ? 8 : 0;
  const int bkb = (lane & 16) ? 8 : 0;
  const bf16_t* Ap = dtlow + ((size_t)tm + (lane & 15)) * DTR + akb;
  const bf16_t* Wp[NT];
#pragma unroll
  for (int i = 0; i < NT; ++i)
    Wp[i] = dtw_bf + ((size_t)(tn0 + i * 16) + (lane & 15)) * DTR + bkb;
  v8f acc[NT];
#pragma unroll
  for (int i = 0; i < NT; ++i) acc[i] = {};
#pragma unroll
  for (int k0 = 0; k0 < DTR; k0 += 32) {
    bf16x16 a = load_frag2(Ap + k0, Ap + k0 + 16);
#pragma unroll
    for (int i = 0; i < NT; ++i) {
      bf16x16 b = load_frag2(Wp[i] + k0, Wp[i] + k0 + 16);
      acc[i] = __builtin_amdgcn_wmma_f32_16x16x32_bf16(false, a, false, b, (short)0,
                                                       acc[i], false, false);
    }
  }
  const int n  = lane & 15;
  const int mb = (lane & 16) ? 8 : 0;
#pragma unroll
  for (int i = 0; i < NT; ++i) {
    const int tn = tn0 + i * 16;
    const float bias = dt_b[tn + n];
#pragma unroll
    for (int r = 0; r < 8; ++r) {
      float v = acc[i][r] + bias;
      delta[(size_t)(tm + mb + r) * DI + tn + n] = softplus_bf(v);  // branch-free
    }
  }
}

// ---------------- selective scan: one thread per (b,d), 16 states in registers ----------------
__global__ void scan_kernel(const float* __restrict__ delta, const float* __restrict__ xc,
                            const float* __restrict__ x_dbl, const float* __restrict__ xz,
                            const float* __restrict__ Anew, const float* __restrict__ Dp,
                            bf16_t* __restrict__ ybf) {
  const int d = blockIdx.x * blockDim.x + threadIdx.x;   // 0..2047
  const int b = blockIdx.y;
  float s[DST];
  float Arow[DST];
#pragma unroll
  for (int n = 0; n < DST; ++n) { s[n] = 0.f; Arow[n] = Anew[d * DST + n]; }
  const float Dd = Dp[d];
  for (int l = 0; l < SEQ; ++l) {
    const size_t bl = (size_t)b * SEQ + l;
    const float dt = delta[bl * DI + d];
    const float u  = xc[bl * DI + d];
    const float* row = x_dbl + bl * XDE;     // [0:64)=dt_low, [64:80)=B, [80:96)=C
    const float du = dt * u;
    float y = 0.f;
#pragma unroll
    for (int n = 0; n < DST; ++n) {
      s[n] = s[n] * __expf(dt * Arow[n]) + du * row[DTR + n];
      y += s[n] * row[DTR + DST + n];
    }
    y += u * Dd;
    const float z = xz[bl * DI2 + DI + d];
    y *= z / (1.f + __expf(-z));             // gate with SiLU(z)
    ybf[bl * DI + d] = (bf16_t)y;
  }
}

// ---------------- host side ----------------
extern "C" void kernel_launch(void* const* d_in, const int* in_sizes, int n_in,
                              void* d_out, int out_size, void* d_ws, size_t ws_size,
                              hipStream_t stream) {
  (void)in_sizes; (void)n_in; (void)out_size; (void)ws_size;
  const float* hidden     = (const float*)d_in[0];
  // d_in[1..3] = T, H, W scalars (compile-time constants here)
  const float* in_proj_w  = (const float*)d_in[4];
  const float* conv_w     = (const float*)d_in[5];
  const float* conv_b     = (const float*)d_in[6];
  const float* x_proj_w   = (const float*)d_in[7];
  const float* dt_proj_w  = (const float*)d_in[8];
  const float* dt_proj_b  = (const float*)d_in[9];
  const float* A_log      = (const float*)d_in[10];
  const float* Dp         = (const float*)d_in[11];
  const float* out_proj_w = (const float*)d_in[12];
  float* out = (float*)d_out;

  char* ws = (char*)d_ws;
  size_t o = 0;
  auto alloc = [&](size_t bytes) { size_t r = o; o = (o + bytes + 255) & ~(size_t)255; return r; };
  bf16_t* hid_bf   = (bf16_t*)(ws + alloc((size_t)BL * DM * 2));
  bf16_t* inw_bf   = (bf16_t*)(ws + alloc((size_t)DI2 * DM * 2));
  bf16_t* xpw_bf   = (bf16_t*)(ws + alloc((size_t)XDE * DI * 2));
  bf16_t* dtw_bf   = (bf16_t*)(ws + alloc((size_t)DI * DTR * 2));
  bf16_t* opw_bf   = (bf16_t*)(ws + alloc((size_t)DM * DI * 2));
  float*  xz       = (float*)(ws + alloc((size_t)BL * DI2 * 4));
  float*  xc       = (float*)(ws + alloc((size_t)BL * DI * 4));
  bf16_t* xc_bf    = (bf16_t*)(ws + alloc((size_t)BL * DI * 2));
  float*  x_dbl    = (float*)(ws + alloc((size_t)BL * XDE * 4));
  bf16_t* dtlow_bf = (bf16_t*)(ws + alloc((size_t)BL * DTR * 2));
  float*  delta    = (float*)(ws + alloc((size_t)BL * DI * 4));
  bf16_t* ybf      = (bf16_t*)(ws + alloc((size_t)BL * DI * 2));
  float*  features = (float*)(ws + alloc((size_t)32 * DI * 4));
  float*  sscr     = (float*)(ws + alloc((size_t)3 * DI * 4));
  float*  Anew     = (float*)(ws + alloc((size_t)DI * DST * 4));

  auto cvt = [&](const float* s, bf16_t* dst, size_t n) {
    f32_to_bf16<<<dim3((unsigned)((n + 255) / 256)), dim3(256), 0, stream>>>(s, dst, n);
  };
  cvt(hidden,     hid_bf, (size_t)BL * DM);
  cvt(in_proj_w,  inw_bf, (size_t)DI2 * DM);
  cvt(x_proj_w,   xpw_bf, (size_t)XDE * DI);
  cvt(dt_proj_w,  dtw_bf, (size_t)DI * DTR);
  cvt(out_proj_w, opw_bf, (size_t)DM * DI);

  // xz = hidden @ in_proj_w^T   (8192 x 4096), 4 N-tiles/wave
  {
    int waves = (BL >> 4) * (DI2 >> 6);
    gemm_bf16<4><<<dim3((unsigned)((waves + 7) / 8)), dim3(256), 0, stream>>>(
        hid_bf, inw_bf, xz, BL, DI2, DM);
  }

  // channel alignment (uses x = xz[:, :DI]) -> features, loss, A inversion
  fft_band_kernel<<<dim3(DI, 32), dim3(64), 0, stream>>>(xz, features);
  stats_invertA_kernel<<<dim3(1), dim3(256), 0, stream>>>(features, A_log, Anew,
                                                          out + OUT_ELEMS, sscr);

  // depthwise causal conv + SiLU
  conv_silu_kernel<<<dim3((BL * DI) / 256), dim3(256), 0, stream>>>(xz, conv_w, conv_b,
                                                                    xc, xc_bf);
  // x_dbl = xc @ x_proj_w^T   (8192 x 96), 2 N-tiles/wave
  {
    int waves = (BL >> 4) * (XDE >> 5);
    gemm_bf16<2><<<dim3((unsigned)((waves + 7) / 8)), dim3(256), 0, stream>>>(
        xc_bf, xpw_bf, x_dbl, BL, XDE, DI);
  }

  // pack dt_low, then delta = softplus(dt_low @ dt_proj_w^T + b)
  pack_dtlow<<<dim3((BL * DTR) / 256), dim3(256), 0, stream>>>(x_dbl, dtlow_bf);
  {
    int waves = (BL >> 4) * (DI >> 6);
    dt_gemm_softplus<<<dim3((unsigned)((waves + 7) / 8)), dim3(256), 0, stream>>>(
        dtlow_bf, dtw_bf, dt_proj_b, delta);
  }

  // selective scan + skip + gate -> y (bf16)
  scan_kernel<<<dim3(DI / 256, BSZc), dim3(256), 0, stream>>>(delta, xc, x_dbl, xz,
                                                              Anew, Dp, ybf);

  // out = y @ out_proj_w^T   (8192 x 1024), 4 N-tiles/wave
  {
    int waves = (BL >> 4) * (DM >> 6);
    gemm_bf16<4><<<dim3((unsigned)((waves + 7) / 8)), dim3(256), 0, stream>>>(
        ybf, opw_bf, out, BL, DM, DI);
  }
}
